// TwoLayerLSH_11536282157422
// MI455X (gfx1250) — compile-verified, matrix operand order checked by reading, not run on hardware
//
#include <hip/hip_runtime.h>

// Problem constants (from reference)
enum { N_ = 2048, D_ = 1024, H_ = 131072, C_ = 1024, S_ = 8192 };

#define DEV __device__ __forceinline__

typedef __attribute__((ext_vector_type(16))) __bf16 v16bf;
typedef __attribute__((ext_vector_type(8)))  float  v8f;
typedef int vi4 __attribute__((vector_size(16)));   // 128-bit payload type of async-LDS builtin

#define AS1 __attribute__((address_space(1)))
#define AS3 __attribute__((address_space(3)))

#if defined(__gfx1250__) && __has_builtin(__builtin_amdgcn_global_load_async_to_lds_b128)
#define HAVE_ASYNC_LDS 1
#else
#define HAVE_ASYNC_LDS 0
#endif

union FragU { v16bf v; uint4 q[2]; };

DEV unsigned short f2bf(float x) {
  union { float f; unsigned int u; } c; c.f = x;
  unsigned int r = c.u + 0x7FFFu + ((c.u >> 16) & 1u);   // round-to-nearest-even
  return (unsigned short)(r >> 16);
}

// Load one 16x32 bf16 WMMA A/B fragment from a row-major [rows, ld] bf16 buffer
// (global or LDS; for LDS these lower to ds_load_b128).
// ISA 16-bit A-matrix 16x32 layout (wave32):
//   lanes 0-15 : M=lane,    VGPR0-3 = K 0..7,  VGPR4-7 = K 16..23
//   lanes 16-31: M=lane-16, VGPR0-3 = K 8..15, VGPR4-7 = K 24..31
DEV v16bf load_frag_nt(const unsigned short* __restrict__ base, int ld) {
  int lane = threadIdx.x & 31;
  int r    = lane & 15;
  int hh   = lane >> 4;
  const unsigned short* p = base + (size_t)r * (size_t)ld + (size_t)(hh * 8);
  FragU f;
  f.q[0] = *(const uint4*)(p);
  f.q[1] = *(const uint4*)(p + 16);
  return f.v;
}

template <int Nw>
DEV void wait_asynccnt_le() {
#if defined(__gfx1250__)
#if __has_builtin(__builtin_amdgcn_s_wait_asynccnt)
  __builtin_amdgcn_s_wait_asynccnt(Nw);
#else
  if (Nw == 0) asm volatile("s_wait_asynccnt 0x0" ::: "memory");
  else         asm volatile("s_wait_asynccnt 0x8" ::: "memory");
#endif
#endif
}

#if HAVE_ASYNC_LDS
// Async-copy one 32-row slice (rows 32w..32w+31) of a 128x32 bf16 tile into LDS
// with rows padded to 40 elements (80 B -> bank-conflict-free ds_load_b128).
// 4 x GLOBAL_LOAD_ASYNC_TO_LDS_B128 per wave (each covers 8 rows: 4 lanes x 16B/row).
DEV void stage_tile(const unsigned short* __restrict__ G, int grow0, int K, int k,
                    unsigned short* ldsbuf, int w, int lane) {
  const int r4 = lane >> 2;        // 0..7: row within 8-row group
  const int c8 = (lane & 3) * 8;   // element column offset (16B granule)
#pragma unroll
  for (int t = 0; t < 4; ++t) {
    const int r = 32 * w + t * 8 + r4;
    const unsigned short* gp = G + (size_t)(grow0 + r) * (size_t)K + k + c8;
    unsigned short* lp = ldsbuf + r * 40 + c8;
    __builtin_amdgcn_global_load_async_to_lds_b128(
        (AS1 vi4*)(unsigned long long)gp,
        (AS3 vi4*)(unsigned int)(unsigned long long)lp,
        0, 0);
  }
}
#endif

// C[M,Nc] = A[M,K] * B[Nc,K]^T + bias, optional ReLU + bf16 output.
// Block = 128 threads = 4 waves in 2x2 -> 128x128 block tile, 64x64 per wave.
template <bool RELU_BF16_OUT>
__global__ __launch_bounds__(128)
void wmma_gemm_nt(const unsigned short* __restrict__ A,
                  const unsigned short* __restrict__ B,
                  const float* __restrict__ bias,
                  void* __restrict__ Cout,
                  int Ncols, int K)
{
  const int w    = threadIdx.x >> 5;
  const int lane = threadIdx.x & 31;
  const int gm   = blockIdx.y * 128;
  const int gn   = blockIdx.x * 128;
  const int m0   = gm + (w >> 1) * 64;
  const int n0   = gn + (w & 1) * 64;

  v8f acc[4][4];
  const v8f zero = {0.f, 0.f, 0.f, 0.f, 0.f, 0.f, 0.f, 0.f};
#pragma unroll
  for (int i = 0; i < 4; ++i)
#pragma unroll
    for (int j = 0; j < 4; ++j) acc[i][j] = zero;

#if HAVE_ASYNC_LDS
  // Double-buffered LDS staging via the gfx1250 async copy engine (ASYNCcnt).
  __shared__ __attribute__((aligned(16))) unsigned short lA[2][128 * 40];
  __shared__ __attribute__((aligned(16))) unsigned short lB[2][128 * 40];

  const int nk = K >> 5;
  stage_tile(A, gm, K, 0, lA[0], w, lane);
  stage_tile(B, gn, K, 0, lB[0], w, lane);
  for (int it = 0; it < nk; ++it) {
    if (it + 1 < nk) {
      stage_tile(A, gm, K, (it + 1) << 5, lA[(it + 1) & 1], w, lane);
      stage_tile(B, gn, K, (it + 1) << 5, lB[(it + 1) & 1], w, lane);
      wait_asynccnt_le<8>();   // 8 newest = prefetch batch; batch 'it' retired (in-order)
    } else {
      wait_asynccnt_le<0>();
    }
    __syncthreads();           // all waves' slices visible
    const unsigned short* sA = lA[it & 1] + ((w >> 1) * 64) * 40;
    const unsigned short* sB = lB[it & 1] + ((w & 1) * 64) * 40;
    v16bf af[4], bfr[4];
#pragma unroll
    for (int i = 0; i < 4; ++i) af[i]  = load_frag_nt(sA + (16 * i) * 40, 40);
#pragma unroll
    for (int j = 0; j < 4; ++j) bfr[j] = load_frag_nt(sB + (16 * j) * 40, 40);
#pragma unroll
    for (int i = 0; i < 4; ++i)
#pragma unroll
      for (int j = 0; j < 4; ++j)
        acc[i][j] = __builtin_amdgcn_wmma_f32_16x16x32_bf16(
            false, af[i], false, bfr[j], (short)0, acc[i][j], false, false);
    __syncthreads();           // protect buffer reuse by next prefetch
  }
#else
  // Fallback: direct global->VGPR fragments (known-good path).
  for (int k = 0; k < K; k += 32) {
    v16bf af[4], bfr[4];
#pragma unroll
    for (int i = 0; i < 4; ++i)
      af[i] = load_frag_nt(A + (size_t)(m0 + 16 * i) * (size_t)K + k, K);
#pragma unroll
    for (int j = 0; j < 4; ++j)
      bfr[j] = load_frag_nt(B + (size_t)(n0 + 16 * j) * (size_t)K + k, K);
#pragma unroll
    for (int i = 0; i < 4; ++i)
#pragma unroll
      for (int j = 0; j < 4; ++j)
        acc[i][j] = __builtin_amdgcn_wmma_f32_16x16x32_bf16(
            false, af[i], false, bfr[j], (short)0, acc[i][j], false, false);
  }
#endif

  // Epilogue. C/D 16x16 f32 layout: lane<16 -> (M=t, N=lane); lane>=16 -> (M=t+8, N=lane-16)
  const int cn = lane & 15;
  const int rh = (lane >> 4) * 8;
#pragma unroll
  for (int j = 0; j < 4; ++j) {
    const int   col = n0 + 16 * j + cn;
    const float bv  = bias[col];
#pragma unroll
    for (int i = 0; i < 4; ++i) {
#pragma unroll
      for (int t = 0; t < 8; ++t) {
        const int row = m0 + 16 * i + rh + t;
        float v = acc[i][j][t] + bv;
        if (RELU_BF16_OUT) {
          v = v > 0.f ? v : 0.f;
          ((unsigned short*)Cout)[(size_t)row * Ncols + col] = f2bf(v);
        } else {
          ((float*)Cout)[(size_t)row * Ncols + col] = v;
        }
      }
    }
  }
}

__global__ __launch_bounds__(256)
void convert_x_bf16(const float* __restrict__ X, unsigned short* __restrict__ Xb, int total) {
  int i = blockIdx.x * blockDim.x + threadIdx.x;
  if (i < total) Xb[i] = f2bf(X[i]);
}

// One block per sampled hidden neuron: copy/convert its weight row (coalesced) + bias.
__global__ __launch_bounds__(256)
void gather_whidden_rows(const float* __restrict__ Wh, const float* __restrict__ bh,
                         const long long* __restrict__ ids,
                         unsigned short* __restrict__ Wg, float* __restrict__ bg) {
  const int s = blockIdx.x;
  const long long id = ids[s];
  const float* src = Wh + (size_t)id * D_;
  unsigned short* dst = Wg + (size_t)s * D_;
  for (int d = threadIdx.x; d < D_; d += blockDim.x) dst[d] = f2bf(src[d]);
  if (threadIdx.x == 0) bg[s] = bh[id];
}

// One block per output class row: gather 8192 random columns (ids staged in LDS).
__global__ __launch_bounds__(256)
void gather_wout_cols(const float* __restrict__ Wo, const long long* __restrict__ ids,
                      unsigned short* __restrict__ Wog) {
  __shared__ int sid[S_];
  for (int s = threadIdx.x; s < S_; s += blockDim.x) sid[s] = (int)ids[s];
  __syncthreads();
  const int c = blockIdx.x;
  const float* src = Wo + (size_t)c * (size_t)H_;
  unsigned short* dst = Wog + (size_t)c * (size_t)S_;
  for (int s = threadIdx.x; s < S_; s += blockDim.x) dst[s] = f2bf(src[sid[s]]);
}

// Workspace layout (bytes): total 69 MB
static const size_t OFF_XB  = 0;                    //  4 MB  X bf16 [N,D]
static const size_t OFF_WG  = (size_t)4  << 20;     // 16 MB  gathered W_hidden bf16 [S,D]
static const size_t OFF_BH  = (size_t)20 << 20;     // 32 KB  gathered b_hidden f32 [S]
static const size_t OFF_WOG = (size_t)21 << 20;     // 16 MB  gathered W_out bf16 [C,S]
static const size_t OFF_H   = (size_t)37 << 20;     // 32 MB  hidden activations bf16 [N,S]

extern "C" void kernel_launch(void* const* d_in, const int* in_sizes, int n_in,
                              void* d_out, int out_size, void* d_ws, size_t ws_size,
                              hipStream_t stream) {
  (void)in_sizes; (void)n_in; (void)out_size; (void)ws_size;
  const float*     X   = (const float*)d_in[0];
  const float*     Wh  = (const float*)d_in[1];
  const float*     bh  = (const float*)d_in[2];
  const float*     Wo  = (const float*)d_in[3];
  const float*     bo  = (const float*)d_in[4];
  const long long* ids = (const long long*)d_in[5];   // int64 sample_ids

  char* ws = (char*)d_ws;
  unsigned short* Xb  = (unsigned short*)(ws + OFF_XB);
  unsigned short* Wg  = (unsigned short*)(ws + OFF_WG);
  float*          bg  = (float*)         (ws + OFF_BH);
  unsigned short* Wog = (unsigned short*)(ws + OFF_WOG);
  unsigned short* Hb  = (unsigned short*)(ws + OFF_H);

  // 1) X -> bf16
  convert_x_bf16<<<(N_ * D_ + 255) / 256, 256, 0, stream>>>(X, Xb, N_ * D_);
  // 2) gather W_hidden rows + bias -> bf16 (coalesced row copies)
  gather_whidden_rows<<<S_, 256, 0, stream>>>(Wh, bh, ids, Wg, bg);
  // 3) gather W_out columns -> bf16 (random gather, ids in LDS)
  gather_wout_cols<<<C_, 256, 0, stream>>>(Wo, ids, Wog);
  // 4) h = relu(X @ Wg^T + bg)  : M=2048, N=8192, K=1024 -> bf16
  wmma_gemm_nt<true><<<dim3(S_ / 128, N_ / 128), 128, 0, stream>>>(
      Xb, Wg, bg, (void*)Hb, S_, D_);
  // 5) out = h @ Wog^T + b_out : M=2048, N=1024, K=8192 -> f32
  wmma_gemm_nt<false><<<dim3(C_ / 128, N_ / 128), 128, 0, stream>>>(
      Hb, Wog, bo, d_out, C_, S_);
}
